// GaussianBlurDM_11184094839260
// MI455X (gfx1250) — matrix operand order repaired and checked by reading, not compiled, not run on hardware
//
#include <hip/hip_runtime.h>
#include <hip/hip_bf16.h>
#include <math.h>

// ---------------------------------------------------------------------------
// GaussianBlurDM on MI455X (gfx1250):
//   t-fold separable Gaussian blur == Z_t = P * Z * P^T with P = B^t,
//   B = 256x256 reflect-padded 11-tap blur matrix.
// Pipeline:
//   1) build_B              : B (f32) in ws
//   2) wmma_mm_f32  x18     : P_s = B * P_{s-1}     (V_WMMA_F32_16X16X4_F32)
//   3) cvt_f32_f16          : powers -> f16, x -> f16
//   4) wmma_gemm_abt_lds<f16>: Yt[b,c]  = P_t * Z[b,c]^T  (V_WMMA_F32_16X16X32_F16)
//   5) wmma_gemm_abt_lds<f32>: Out[b,c] = P_t * Yt[b,c]^T -> d_out (f32)
// Main GEMMs: 128x128 block tile, 64x32 wave tile (8 accumulators/wave),
// A/B panels staged to LDS via GLOBAL_LOAD_ASYNC_TO_LDS_B128 (ASYNCcnt).
// ---------------------------------------------------------------------------

typedef __attribute__((ext_vector_type(16))) _Float16 v16h;
typedef __attribute__((ext_vector_type(8)))  _Float16 v8h;
typedef __attribute__((ext_vector_type(8)))  float    v8f;
typedef __attribute__((ext_vector_type(2)))  float    v2f;
typedef int v4i_ __attribute__((vector_size(16)));

#define NDIM 256
#define KTAP 11
#define PADR 5
#define NPOW 19
#define NPLANE 192   // 64 batches * 3 channels

// Main GEMM blocking
#define BM 128
#define BN 128
#define BK 32
#define LPITCH 40    // halves per LDS row (80 B): every 16B-frag read stays aligned

__device__ __forceinline__ int iclamp(int v, int lo, int hi) {
    return v < lo ? lo : (v > hi ? hi : v);
}

// ---- CDNA5 async global->LDS staging (16 B per call, per lane) -------------
__device__ __forceinline__ void stage16(const _Float16* g, _Float16* l) {
#if defined(__gfx1250__) && __has_builtin(__builtin_amdgcn_global_load_async_to_lds_b128)
    __builtin_amdgcn_global_load_async_to_lds_b128(
        (__attribute__((address_space(1))) v4i_*)g,
        (__attribute__((address_space(3))) v4i_*)l, 0, 0);
#else
    *(v8h*)l = *(const v8h*)g;   // fallback: vmem load + ds_store
#endif
}

__device__ __forceinline__ void wait_async() {
#if defined(__gfx1250__)
#if __has_builtin(__builtin_amdgcn_s_wait_asynccnt)
    __builtin_amdgcn_s_wait_asynccnt(0);
#else
    asm volatile("s_wait_asynccnt 0x0" ::: "memory");
#endif
#endif
}

// ---------------------------------------------------------------------------
// Build the 256x256 reflect-blur matrix B. One thread per row.
// ---------------------------------------------------------------------------
__global__ void build_B(float* __restrict__ Bm) {
    const int i = threadIdx.x;            // row index, blockDim.x == 256
    float w[KTAP];
    float s = 0.0f;
    #pragma unroll
    for (int k = 0; k < KTAP; ++k) {
        float xk = (float)k - 5.0f;
        w[k] = __expf(-0.125f * xk * xk);   // sigma=2 -> 0.5/sigma^2 = 0.125
        s += w[k];
    }
    const float inv = 1.0f / s;

    float* row = Bm + i * NDIM;
    for (int j = 0; j < NDIM; ++j) row[j] = 0.0f;
    #pragma unroll
    for (int k = 0; k < KTAP; ++k) {
        int idx = i + k - PADR;
        if (idx < 0) idx = -idx;                       // 'reflect' mode
        if (idx > NDIM - 1) idx = 2 * (NDIM - 1) - idx;
        row[idx] += w[k] * inv;                        // same thread: no race
    }
}

// ---------------------------------------------------------------------------
// C = A * Bm, 256x256 f32, via V_WMMA_F32_16X16X4_F32 (power chain; tiny,
// L2-resident, off the critical path).
// ---------------------------------------------------------------------------
__global__ void wmma_mm_f32(const float* __restrict__ A,
                            const float* __restrict__ Bm,
                            float* __restrict__ C) {
    const int lane = threadIdx.x & 31;
    const int wave = threadIdx.x >> 5;
    const int tile = blockIdx.x * 8 + wave;       // 0..255
    const int i0 = (tile >> 4) << 4;
    const int j0 = (tile & 15) << 4;
    const int lr = lane & 15;
    const int lh = lane >> 4;
    const int ka = lh * 2;

    const float* arow = A  + (i0 + lr) * NDIM;
    const float* bcol = Bm + (j0 + lr);

    v8f acc = {};
    for (int k0 = 0; k0 < NDIM; k0 += 4) {
        v2f a, b;
        a[0] = arow[k0 + ka];
        a[1] = arow[k0 + ka + 1];
        b[0] = bcol[(k0 + ka) * NDIM];
        b[1] = bcol[(k0 + ka + 1) * NDIM];
        acc = __builtin_amdgcn_wmma_f32_16x16x4_f32(
            false, a, false, b, (short)0, acc, false, false);
    }

    float* crow = C + (i0 + lh * 8) * NDIM + j0 + lr;
    #pragma unroll
    for (int r = 0; r < 8; ++r) crow[r * NDIM] = acc[r];
}

// ---------------------------------------------------------------------------
// Elementwise f32 -> f16
// ---------------------------------------------------------------------------
__global__ void cvt_f32_f16(const float* __restrict__ in,
                            _Float16* __restrict__ out, int n) {
    int i = blockIdx.x * blockDim.x + threadIdx.x;
    if (i < n) out[i] = (_Float16)in[i];
}

// ---------------------------------------------------------------------------
// C[plane] = A_t * Bsrc[plane]^T, f16 in / f32 accumulate.
// Block (256 thr, 8 waves) -> 128x128 tile; wave -> 64x32 (8 acc tiles).
// Per BK=32 chunk: async-stage A/B 128x32 panels to LDS, then per wave
// 4 A-frags + 2 B-frags (aligned ds_load_b128) feed 8 WMMAs.
// ---------------------------------------------------------------------------
template <bool OUT_F16>
__global__ void __launch_bounds__(256)
wmma_gemm_abt_lds(const _Float16* __restrict__ Pows,  // [19][256*256]
                  const int* __restrict__ tarr,       // [64]
                  const _Float16* __restrict__ Bsrc,  // [192][256*256]
                  void* __restrict__ OutP) {
    __shared__ __align__(16) _Float16 Alds[BM * LPITCH];
    __shared__ __align__(16) _Float16 Blds[BN * LPITCH];

    const int plane = blockIdx.x >> 2;            // 4 blocks per plane (2x2)
    const int bb    = blockIdx.x & 3;
    const int i0 = (bb >> 1) * BM;
    const int j0 = (bb & 1) * BN;

    const int tid  = threadIdx.x;
    const int lane = tid & 31;
    const int wave = tid >> 5;
    const int lr = lane & 15;
    const int lh = lane >> 4;
    const int wr = wave >> 2;                     // 0..1 -> M offset (64)
    const int wc = wave & 3;                      // 0..3 -> N offset (32)

    const int t = iclamp(tarr[plane / 3], 1, NPOW) - 1;
    const _Float16* Ag = Pows + (size_t)t * (NDIM * NDIM);
    const _Float16* Bg = Bsrc + (size_t)plane * (NDIM * NDIM);

    // staging: each thread moves 32 B per panel per chunk (2x 16B async ops)
    const int srow = tid >> 1;                    // 0..127
    const int skh  = (tid & 1) * 16;              // halves

    v8f acc[4][2] = {};

    for (int k0 = 0; k0 < NDIM; k0 += BK) {
        const _Float16* ga = Ag + (size_t)(i0 + srow) * NDIM + k0 + skh;
        const _Float16* gb = Bg + (size_t)(j0 + srow) * NDIM + k0 + skh;
        _Float16* la = Alds + srow * LPITCH + skh;
        _Float16* lb = Blds + srow * LPITCH + skh;
        stage16(ga,     la);
        stage16(ga + 8, la + 8);
        stage16(gb,     lb);
        stage16(gb + 8, lb + 8);
        wait_async();
        __syncthreads();

        v16h af[4], bf[2];
        #pragma unroll
        for (int mt = 0; mt < 4; ++mt) {
            const _Float16* ar = Alds + (wr * 64 + mt * 16 + lr) * LPITCH;
            ((v8h*)&af[mt])[0] = *(const v8h*)(ar + lh * 8);        // K 0-7 / 8-15
            ((v8h*)&af[mt])[1] = *(const v8h*)(ar + 16 + lh * 8);   // K 16-23 / 24-31
        }
        #pragma unroll
        for (int nt = 0; nt < 2; ++nt) {
            const _Float16* br = Blds + (wc * 32 + nt * 16 + lr) * LPITCH;
            ((v8h*)&bf[nt])[0] = *(const v8h*)(br + lh * 8);        // K 0-7 / 8-15
            ((v8h*)&bf[nt])[1] = *(const v8h*)(br + 16 + lh * 8);   // K 16-23 / 24-31
        }
        #pragma unroll
        for (int mt = 0; mt < 4; ++mt)
            #pragma unroll
            for (int nt = 0; nt < 2; ++nt)
                acc[mt][nt] = __builtin_amdgcn_wmma_f32_16x16x32_f16(
                    false, af[mt], false, bf[nt], (short)0, acc[mt][nt],
                    false, false);
        __syncthreads();
    }

    #pragma unroll
    for (int mt = 0; mt < 4; ++mt) {
        #pragma unroll
        for (int nt = 0; nt < 2; ++nt) {
            const int row = i0 + wr * 64 + mt * 16 + lh * 8;
            const int col = j0 + wc * 32 + nt * 16 + lr;
            if (OUT_F16) {
                _Float16* o = (_Float16*)OutP + (size_t)plane * (NDIM * NDIM)
                              + (size_t)row * NDIM + col;
                #pragma unroll
                for (int r = 0; r < 8; ++r) o[r * NDIM] = (_Float16)acc[mt][nt][r];
            } else {
                float* o = (float*)OutP + (size_t)plane * (NDIM * NDIM)
                           + (size_t)row * NDIM + col;
                #pragma unroll
                for (int r = 0; r < 8; ++r) o[r * NDIM] = acc[mt][nt][r];
            }
        }
    }
}

// ---------------------------------------------------------------------------
// Host launcher. Workspace layout (bytes):
//   [0, 4.75MB)          P_f32[19]  (P_f32[0] = B)
//   [+ , +2.375MB)       P_f16[19]
//   [+ , +25.2MB)        xh  = x as f16      (192 planes)
//   [+ , +25.2MB)        Yt  = P * Z^T (f16) (192 planes)
// Total ~57.5 MB, all L2-resident on MI455X (192 MB L2).
// ---------------------------------------------------------------------------
extern "C" void kernel_launch(void* const* d_in, const int* in_sizes, int n_in,
                              void* d_out, int out_size, void* d_ws, size_t ws_size,
                              hipStream_t stream) {
    const float* x = (const float*)d_in[0];
    const int*   t = (const int*)d_in[1];
    float*     out = (float*)d_out;

    const size_t MAT = (size_t)NDIM * NDIM;
    char* ws = (char*)d_ws;
    float*    Pf32 = (float*)ws;
    _Float16* Pf16 = (_Float16*)(ws + NPOW * MAT * sizeof(float));
    _Float16* xh   = (_Float16*)((char*)Pf16 + NPOW * MAT * sizeof(_Float16));
    _Float16* Yt   = xh + NPLANE * MAT;

    // 1) B
    build_B<<<1, 256, 0, stream>>>(Pf32);

    // 2) powers P_s = B * P_{s-1}, s = 2..19 (18 small dependent f32 WMMA GEMMs)
    for (int s = 1; s < NPOW; ++s) {
        wmma_mm_f32<<<32, 256, 0, stream>>>(Pf32, Pf32 + (size_t)(s - 1) * MAT,
                                            Pf32 + (size_t)s * MAT);
    }

    // 3) convert powers and input to f16
    {
        int n = (int)(NPOW * MAT);
        cvt_f32_f16<<<(n + 255) / 256, 256, 0, stream>>>(Pf32, Pf16, n);
    }
    {
        int n = (int)(NPLANE * MAT);
        cvt_f32_f16<<<(n + 255) / 256, 256, 0, stream>>>(x, xh, n);
    }

    // 4) Yt[b,c] = P_t * Z[b,c]^T       (f16 out)
    wmma_gemm_abt_lds<true><<<NPLANE * 4, 256, 0, stream>>>(Pf16, t, xh, (void*)Yt);

    // 5) Out[b,c] = P_t * Yt[b,c]^T     (f32 out -> d_out)
    wmma_gemm_abt_lds<false><<<NPLANE * 4, 256, 0, stream>>>(Pf16, t, Yt, (void*)out);
}
